// my_selfattention1_76441827934626
// MI455X (gfx1250) — matrix operand order repaired and checked by reading, not compiled, not run on hardware
//
#include <hip/hip_runtime.h>

// ---------- types ----------
typedef __attribute__((ext_vector_type(16))) _Float16 v16h;
typedef __attribute__((ext_vector_type(8)))  _Float16 v8h;
typedef __attribute__((ext_vector_type(8)))  float    v8f;
typedef __attribute__((ext_vector_type(4)))  float    v4f;

// ---------- problem constants ----------
#define E_DIM   512
#define BATCH   1024
#define HEADS   8
#define DH      64
#define SQ      96
#define SK      8
#define S_TOT   104
#define MQ      (SQ * BATCH)      // 98304
#define MK      (SK * BATCH)      // 8192

// ---------------------------------------------------------------
// f32 -> f16 conversion, 8 elements per thread (b128 in/out)
// n must be a multiple of 8 (true for all arrays here)
// ---------------------------------------------------------------
__global__ void cvt_f32_to_f16_x8(const float* __restrict__ in,
                                  _Float16* __restrict__ out, long long n8) {
    long long i = (long long)blockIdx.x * blockDim.x + threadIdx.x;
    long long stride = (long long)gridDim.x * blockDim.x;
    for (; i < n8; i += stride) {
        v4f a = *(const v4f*)(in + i * 8);
        v4f b = *(const v4f*)(in + i * 8 + 4);
        v8h h;
#pragma unroll
        for (int e = 0; e < 4; ++e) { h[e] = (_Float16)a[e]; h[e + 4] = (_Float16)b[e]; }
        *(v8h*)(out + i * 8) = h;
    }
}

// ---------------------------------------------------------------
// WMMA GEMM: C[m,n] = act( sum_k A[m,k] * W[n,k] )
// A: (M,K) f16 row-major, W: (N,K) f16 row-major.
// Block tile 128x128, 256 threads = 8 waves, wave tile 32(M)x64(N).
// Double-buffered LDS (ping-pong): prefetch next 32-wide K slice from
// global into VGPRs while WMMAs consume the current buffer.
// LDS row stride 40 halves (80 B) -> conflict-free ds_load_b128 frags.
// M,N multiples of 128; K multiple of 32 (all true here).
// ---------------------------------------------------------------
#define LDS_STRIDE 40
#define LDS_BUF    (128 * LDS_STRIDE)   // 5120 halves per buffer

template <bool PRELU, bool OUTF16>
__global__ __launch_bounds__(256) void gemm_wmma(
    const _Float16* __restrict__ A, const _Float16* __restrict__ W,
    void* __restrict__ Cout, int M, int N, int K,
    const float* __restrict__ alphap) {

    __shared__ _Float16 As[2 * LDS_BUF];   // 20 KB
    __shared__ _Float16 Bs[2 * LDS_BUF];   // 20 KB

    const int tid  = threadIdx.x;
    const int wave = tid >> 5;
    const int lane = tid & 31;
    const int l15  = lane & 15;
    const int hi   = lane >> 4;

    const int bm = blockIdx.x * 128;
    const int bn = blockIdx.y * 128;
    const int wm = (wave & 3) * 32;   // wave M offset in block tile
    const int wn = (wave >> 2) * 64;  // wave N offset in block tile

    const float alpha = PRELU ? (float)*alphap : 0.0f;

    // ---- staging: each thread owns 2 chunks (rows r, r+64) of A and B ----
    const int srow = tid >> 2;              // 0..63
    const int soff = (tid & 3) * 8;         // 0,8,16,24
    const _Float16* pA0 = A + (size_t)(bm + srow) * K + soff;
    const _Float16* pA1 = pA0 + (size_t)64 * K;
    const _Float16* pB0 = W + (size_t)(bn + srow) * K + soff;
    const _Float16* pB1 = pB0 + (size_t)64 * K;
    const int sl0 = srow * LDS_STRIDE + soff;        // LDS chunk offsets
    const int sl1 = sl0 + 64 * LDS_STRIDE;

    // ---- fragment LDS offsets (loop-invariant; + buffer toggle) ----
    int aoff[2][2], boff[4][2];
#pragma unroll
    for (int i = 0; i < 2; ++i) {
        int base = (wm + i * 16 + l15) * LDS_STRIDE;
        aoff[i][0] = base + hi * 8;         // K in {hi*8 + 0..7}
        aoff[i][1] = base + 16 + hi * 8;    // K in {16 + hi*8 + 0..7}
    }
#pragma unroll
    for (int j = 0; j < 4; ++j) {
        int base = (wn + j * 16 + l15) * LDS_STRIDE;
        boff[j][0] = base + hi * 8;         // K in {hi*8 + 0..7}
        boff[j][1] = base + 16 + hi * 8;    // K in {16 + hi*8 + 0..7}
    }

    v8f acc[2][4] = {};

    // ---- prologue: stage K-slice 0 into buffer 0 ----
    *(v8h*)&As[sl0] = *(const v8h*)pA0;
    *(v8h*)&As[sl1] = *(const v8h*)pA1;
    *(v8h*)&Bs[sl0] = *(const v8h*)pB0;
    *(v8h*)&Bs[sl1] = *(const v8h*)pB1;
    __syncthreads();

    int cur = 0;
    for (int k0 = 0; k0 < K; k0 += 32) {
        pA0 += 32; pA1 += 32; pB0 += 32; pB1 += 32;
        const bool more = (k0 + 32 < K);

        // prefetch next slice into registers (overlaps with WMMAs below)
        v8h na0, na1, nb0, nb1;
        if (more) {
            na0 = *(const v8h*)pA0;
            na1 = *(const v8h*)pA1;
            nb0 = *(const v8h*)pB0;
            nb1 = *(const v8h*)pB1;
        }

        // consume current buffer
        const int cb = cur * LDS_BUF;
        union Frag { v16h v; v8h h[2]; };
        Frag af[2], bf[4];
#pragma unroll
        for (int i = 0; i < 2; ++i) {
            af[i].h[0] = *(const v8h*)&As[cb + aoff[i][0]];
            af[i].h[1] = *(const v8h*)&As[cb + aoff[i][1]];
        }
#pragma unroll
        for (int j = 0; j < 4; ++j) {
            bf[j].h[0] = *(const v8h*)&Bs[cb + boff[j][0]];
            bf[j].h[1] = *(const v8h*)&Bs[cb + boff[j][1]];
        }
#pragma unroll
        for (int i = 0; i < 2; ++i)
#pragma unroll
            for (int j = 0; j < 4; ++j)
                acc[i][j] = __builtin_amdgcn_wmma_f32_16x16x32_f16(
                    false, af[i].v, false, bf[j].v, (short)0, acc[i][j],
                    false, false);

        // store prefetched slice into the other buffer
        if (more) {
            const int nb = (cur ^ 1) * LDS_BUF;
            *(v8h*)&As[nb + sl0] = na0;
            *(v8h*)&As[nb + sl1] = na1;
            *(v8h*)&Bs[nb + sl0] = nb0;
            *(v8h*)&Bs[nb + sl1] = nb1;
        }
        __syncthreads();
        cur ^= 1;
    }

    // C/D layout: lane -> n = l15; VGPR r -> m = 8*hi + r
#pragma unroll
    for (int i = 0; i < 2; ++i)
#pragma unroll
        for (int j = 0; j < 4; ++j) {
            int n  = bn + wn + j * 16 + l15;
            int mb = bm + wm + i * 16 + hi * 8;
#pragma unroll
            for (int r = 0; r < 8; ++r) {
                float x = acc[i][j][r];
                if (PRELU) x = (x >= 0.0f) ? x : alpha * x;
                size_t idx = (size_t)(mb + r) * N + n;
                if (OUTF16) ((_Float16*)Cout)[idx] = (_Float16)x;
                else        ((float*)Cout)[idx]    = x;
            }
        }
}

// ---------------------------------------------------------------
// Attention: one block per batch b (all 8 heads -> local, deterministic
// head-means, no atomics). qh: (SQ*B, E) f16; kvh: (SK*B, 2E) f16.
// Softmax is over the flattened Sq*Sk = 768 entries per head.
// ---------------------------------------------------------------
__global__ __launch_bounds__(256) void attn_kernel(
    const _Float16* __restrict__ qh, const _Float16* __restrict__ kvh,
    _Float16* __restrict__ ctx, float* __restrict__ sm_mean,
    float* __restrict__ a_mean) {

    const int b   = blockIdx.x;   // 0..1023
    const int tid = threadIdx.x;

    __shared__ _Float16 kbuf[SK * E_DIM];      //  8 KB
    __shared__ _Float16 vbuf[SK * E_DIM];      //  8 KB
    __shared__ float    at[HEADS * SQ * SK];   // 24 KB, [h][s][tk]

    // stage k,v for this batch: 8 rows x 512 halves = 512 chunks of 8
    for (int c = tid; c < 512; c += 256) {
        int row = c >> 6;
        int off = (c & 63) * 8;
        size_t g = ((size_t)row * BATCH + b) * (2 * E_DIM) + off;
        *(v8h*)&kbuf[row * E_DIM + off] = *(const v8h*)&kvh[g];
        *(v8h*)&vbuf[row * E_DIM + off] = *(const v8h*)&kvh[g + E_DIM];
    }
    __syncthreads();

    const float scale = 0.044194173824159216f;  // 512^-0.5

    // raw attn: 768 (h,s) pairs; each thread loads its q row once
    for (int p = tid; p < HEADS * SQ; p += 256) {
        int h = p / SQ, s = p % SQ;
        const _Float16* qrow = &qh[((size_t)s * BATCH + b) * E_DIM + h * DH];
        v8h qv[8];
#pragma unroll
        for (int u = 0; u < 8; ++u) qv[u] = *(const v8h*)&qrow[u * 8];
#pragma unroll
        for (int tk = 0; tk < SK; ++tk) {
            const _Float16* krow = &kbuf[tk * E_DIM + h * DH];
            float sum = 0.0f;
#pragma unroll
            for (int u = 0; u < 8; ++u)
#pragma unroll
                for (int e = 0; e < 8; ++e)
                    sum += (float)qv[u][e] * (float)krow[u * 8 + e];
            // attn + where(attn==0, -1e4, 0), then * scale
            if (sum == 0.0f) sum = -10000.0f;
            sum *= scale;
            at[h * (SQ * SK) + s * SK + tk] = sum;
        }
    }
    __syncthreads();

    // attn_mean over heads (pre-softmax, post-mask/scale)
    for (int p = tid; p < SQ * SK; p += 256) {
        float s = 0.0f;
#pragma unroll
        for (int h = 0; h < HEADS; ++h) s += at[h * (SQ * SK) + p];
        a_mean[(size_t)b * (SQ * SK) + p] = s * 0.125f;
    }
    __syncthreads();

    // softmax: wave w owns head w, reduction over 768 via wave32 shuffles
    {
        int h = tid >> 5, lane = tid & 31;
        float* base = &at[h * (SQ * SK)];
        float mx = -3.4e38f;
        for (int i = lane; i < SQ * SK; i += 32) mx = fmaxf(mx, base[i]);
#pragma unroll
        for (int o = 16; o; o >>= 1) mx = fmaxf(mx, __shfl_xor(mx, o, 32));
        float sum = 0.0f;
        for (int i = lane; i < SQ * SK; i += 32) sum += expf(base[i] - mx);
#pragma unroll
        for (int o = 16; o; o >>= 1) sum += __shfl_xor(sum, o, 32);
        float inv = 1.0f / sum;
        for (int i = lane; i < SQ * SK; i += 32)
            base[i] = expf(base[i] - mx) * inv;
    }
    __syncthreads();

    // attn_sm_mean over heads
    for (int p = tid; p < SQ * SK; p += 256) {
        float s = 0.0f;
#pragma unroll
        for (int h = 0; h < HEADS; ++h) s += at[h * (SQ * SK) + p];
        sm_mean[(size_t)b * (SQ * SK) + p] = s * 0.125f;
    }

    // ctx[s, b, e] = sum_tk attn_sm[h][s][tk] * v[tk][e]   (h = e/64)
    for (int o = tid; o < SQ * E_DIM; o += 256) {
        int s = o >> 9, e = o & (E_DIM - 1), h = e >> 6;
        const float* ar = &at[h * (SQ * SK) + s * SK];
        float acc = 0.0f;
#pragma unroll
        for (int tk = 0; tk < SK; ++tk)
            acc += ar[tk] * (float)vbuf[tk * E_DIM + e];
        ctx[((size_t)s * BATCH + b) * E_DIM + e] = (_Float16)acc;
    }
}

// ---------------------------------------------------------------
// launcher
// ---------------------------------------------------------------
extern "C" void kernel_launch(void* const* d_in, const int* in_sizes, int n_in,
                              void* d_out, int out_size, void* d_ws,
                              size_t ws_size, hipStream_t stream) {
    const float* ipt    = (const float*)d_in[0];  // (104,1024,512)
    const float* Wq     = (const float*)d_in[1];  // (512,512)
    const float* Wkv    = (const float*)d_in[2];  // (1024,512)
    const float* Wpost  = (const float*)d_in[3];  // (512,512)
    const float* alphap = (const float*)d_in[4];  // scalar

    // workspace layout (f16 halves)
    const size_t nX     = (size_t)S_TOT * BATCH * E_DIM;   // 54,525,952
    const size_t nWq    = (size_t)E_DIM * E_DIM;
    const size_t nWkv   = (size_t)2 * E_DIM * E_DIM;
    const size_t nWpost = (size_t)E_DIM * E_DIM;
    const size_t nQ     = (size_t)MQ * E_DIM;
    const size_t nKV    = (size_t)MK * 2 * E_DIM;

    _Float16* Xh    = (_Float16*)d_ws;
    _Float16* Wqh   = Xh + nX;
    _Float16* Wkvh  = Wqh + nWq;
    _Float16* Wph   = Wkvh + nWkv;
    _Float16* Qh    = Wph + nWpost;
    _Float16* KVh   = Qh + nQ;
    _Float16* Ctxh  = KVh + nKV;
    (void)ws_size; (void)in_sizes; (void)n_in; (void)out_size;

    float* out_ptr = (float*)d_out;                              // 96*1024*512
    float* sm_mean = out_ptr + (size_t)SQ * BATCH * E_DIM;       // (1024,96,8)
    float* a_mean  = sm_mean + (size_t)BATCH * SQ * SK;          // (1024,96,8)

    // 1) f32 -> f16 conversions (8 elems/thread)
    cvt_f32_to_f16_x8<<<(unsigned)((nX / 8 + 255) / 256), 256, 0, stream>>>(
        ipt, Xh, (long long)(nX / 8));
    cvt_f32_to_f16_x8<<<(unsigned)((nWq / 8 + 255) / 256), 256, 0, stream>>>(
        Wq, Wqh, (long long)(nWq / 8));
    cvt_f32_to_f16_x8<<<(unsigned)((nWkv / 8 + 255) / 256), 256, 0, stream>>>(
        Wkv, Wkvh, (long long)(nWkv / 8));
    cvt_f32_to_f16_x8<<<(unsigned)((nWpost / 8 + 255) / 256), 256, 0, stream>>>(
        Wpost, Wph, (long long)(nWpost / 8));

    // 2) q = prelu(X[0:96] @ Wq^T)   (98304 x 512)
    gemm_wmma<true, true><<<dim3(MQ / 128, E_DIM / 128), 256, 0, stream>>>(
        Xh, Wqh, Qh, MQ, E_DIM, E_DIM, alphap);

    // 3) kv = prelu(X[96:104] @ Wkv^T)  (8192 x 1024)
    gemm_wmma<true, true><<<dim3(MK / 128, (2 * E_DIM) / 128), 256, 0, stream>>>(
        Xh + (size_t)SQ * BATCH * E_DIM, Wkvh, KVh, MK, 2 * E_DIM, E_DIM, alphap);

    // 4) attention + means + ctx (one block per batch)
    attn_kernel<<<BATCH, 256, 0, stream>>>(Qh, KVh, Ctxh, sm_mean, a_mean);

    // 5) out = ctx @ Wpost^T -> fp32 d_out
    gemm_wmma<false, false><<<dim3(MQ / 128, E_DIM / 128), 256, 0, stream>>>(
        Ctxh, Wph, out_ptr, MQ, E_DIM, E_DIM, alphap);
}
